// AttentionBlock_59768764891288
// MI455X (gfx1250) — compile-verified, matrix operand order checked by reading, not run on hardware
//
#include <hip/hip_runtime.h>
#include <hip/hip_fp16.h>

#define DI __device__ __forceinline__

typedef _Float16 v16h __attribute__((ext_vector_type(16)));
typedef _Float16 v8h  __attribute__((ext_vector_type(8)));
typedef float    v8f  __attribute__((ext_vector_type(8)));
typedef unsigned int u32x4 __attribute__((ext_vector_type(4)));
typedef int i32x4 __attribute__((ext_vector_type(4)));
typedef int i32x8 __attribute__((ext_vector_type(8)));

constexpr int B = 4, C = 256, N = 4096;
constexpr int THREADS = 256;
constexpr int JB = 32;  // key-block width in kernel 2

// ---------------- TDM availability ----------------
#if defined(__has_builtin)
#if __has_builtin(__builtin_amdgcn_tensor_load_to_lds)
#define HAVE_TDM 1
#if __clang_major__ >= 23
#define TDM_SIX_ARGS 1
#endif
#endif
#endif

// Workspace layout (bytes), all offsets 256B aligned
constexpr size_t W16Q_OFF = 0;
constexpr size_t W16K_OFF = W16Q_OFF + (size_t)C * C * 2;
constexpr size_t W16V_OFF = W16K_OFF + (size_t)C * C * 2;
constexpr size_t QT16_OFF = W16V_OFF + (size_t)C * C * 2;      // [B][N][C] f16  (row i, col c)
constexpr size_t KT16_OFF = QT16_OFF + (size_t)B * N * C * 2;  // [B][N][C] f16  (row j, col c)
constexpr size_t V16_OFF  = KT16_OFF + (size_t)B * N * C * 2;  // [B][C][N] f16  (row c, col j)

// Dynamic-LDS layout for k_attn (element offsets; dynamic segment starts at LDS offset 0)
constexpr unsigned KBUF_E = 0;                    // [2][JB][C]
constexpr unsigned VBUF_E = KBUF_E + 2 * JB * C;  // [2][C][JB]
constexpr unsigned PBUF_E = VBUF_E + 2 * C * JB;  // [8][16][JB]
constexpr unsigned SMEM_BYTES = (PBUF_E + 8 * 16 * JB) * 2;  // 73728 B

// A/B fragment loader. Per CDNA5 ISA 16-bit A layout (lane m = L%16, hk = L/16):
// elems 0..7  = mat[row][base + 8*hk .. +7], elems 8..15 = mat[row][base + 16 + 8*hk .. +7]
DI v16h load_frag(const _Float16* p, int hk) {
  v8h lo = *(const v8h*)(p + 8 * hk);
  v8h hi = *(const v8h*)(p + 16 + 8 * hk);
  return __builtin_shufflevector(lo, hi, 0, 1, 2, 3, 4, 5, 6, 7, 8, 9, 10, 11, 12, 13, 14, 15);
}

DI v8f wmma_f16(v16h a, v16h b, v8f c) {
  return __builtin_amdgcn_wmma_f32_16x16x32_f16(false, a, false, b, (short)0, c, false, false);
}
// reuse-A hint must be a frontend constant -> template parameter
template <bool RA>
DI v8f wmma_f16_t(v16h a, v16h b, v8f c) {
  return __builtin_amdgcn_wmma_f32_16x16x32_f16(false, a, false, b, (short)0, c, RA, false);
}

#if defined(HAVE_TDM)
// Issue a 2D TDM tile load: tensor is [tensor_d1 rows][tensor_d0 elems/row] of f16 with
// row stride stride0 (elements); tile is [tile_d1 rows][tile_d0 elems], packed into LDS
// at byte offset lds_off. D# bit layout per CDNA5 ISA 08_async_tensor (groups 0/1).
DI void tdm_load_2d(unsigned lds_off, const void* gaddr, unsigned tensor_d0, unsigned tensor_d1,
                    unsigned tile_d0, unsigned tile_d1, unsigned long long stride0) {
  unsigned long long ga = (unsigned long long)(uintptr_t)gaddr;
  u32x4 g0;
  g0[0] = 1u;                                                 // count=1, user mode
  g0[1] = lds_off;                                            // lds_addr -> bits[63:32]
  g0[2] = (unsigned)(ga & 0xFFFFFFFFu);                       // global_addr[31:0]
  g0[3] = (unsigned)((ga >> 32) & 0x1FFFFFFu) | (2u << 30);   // global_addr[56:32] | type=2
  i32x8 g1;
  g1[0] = (int)(1u << 16);                                    // data_size=1 (2 bytes), no mask
  g1[1] = (int)((tensor_d0 & 0xFFFFu) << 16);                 // tensor_dim0[15:0] @ bits 63:48
  g1[2] = (int)((tensor_d0 >> 16) | ((tensor_d1 & 0xFFFFu) << 16));
  g1[3] = (int)((tensor_d1 >> 16) | (tile_d0 << 16));         // tile_dim0 @ bits 127:112
  g1[4] = (int)(tile_d1 & 0xFFFFu);                           // tile_dim1; tile_dim2=0
  g1[5] = (int)(unsigned)(stride0 & 0xFFFFFFFFu);             // tensor_dim0_stride[31:0]
  g1[6] = (int)(unsigned)((stride0 >> 32) & 0xFFFFu);         // stride[47:32]; dim1_stride=0
  g1[7] = 0;
  i32x4 z4 = {0, 0, 0, 0};
#if defined(TDM_SIX_ARGS)
  i32x8 z8 = {0, 0, 0, 0, 0, 0, 0, 0};
  __builtin_amdgcn_tensor_load_to_lds(g0, g1, z4, z4, z8, 0);
#else
  __builtin_amdgcn_tensor_load_to_lds(g0, g1, z4, z4, 0);
#endif
}
#endif  // HAVE_TDM

// ---------------- Kernel 0: fp32 weights -> f16 ----------------
__global__ void k_cvt_w(const float* __restrict__ wq, const float* __restrict__ wk,
                        const float* __restrict__ wv, _Float16* __restrict__ w16) {
  int i = blockIdx.x * THREADS + threadIdx.x;
  if (i >= 3 * C * C) return;
  int which = i / (C * C), j = i % (C * C);
  const float* src = which == 0 ? wq : (which == 1 ? wk : wv);
  w16[(size_t)which * C * C + j] = (_Float16)src[j];
}

// ---------------- Kernel 1: QKV projection via WMMA ----------------
__global__ __launch_bounds__(THREADS) void k_qkv(
    const float* __restrict__ x, const float* __restrict__ bq, const float* __restrict__ bk,
    const float* __restrict__ bv, const _Float16* __restrict__ w16q,
    const _Float16* __restrict__ w16k, const _Float16* __restrict__ w16v,
    _Float16* __restrict__ qT, _Float16* __restrict__ kT, _Float16* __restrict__ v16) {
  __shared__ alignas(16) _Float16 xT[64][C];  // 32 KB
  const int tid = threadIdx.x;
  const int b = blockIdx.x / (N / 64);
  const int n0 = (blockIdx.x % (N / 64)) * 64;
  const int wave = tid >> 5, lane = tid & 31, m16 = lane & 15, hk = lane >> 4;

#pragma unroll
  for (int t = 0; t < 16; ++t) {
    int id = tid + THREADS * t;  // 0..4095
    int c = id >> 4, nq = id & 15;
    const float4 xv = *(const float4*)(x + ((size_t)b * C + c) * N + n0 + nq * 4);
    xT[nq * 4 + 0][c] = (_Float16)xv.x;
    xT[nq * 4 + 1][c] = (_Float16)xv.y;
    xT[nq * 4 + 2][c] = (_Float16)xv.z;
    xT[nq * 4 + 3][c] = (_Float16)xv.w;
  }
  __syncthreads();

  for (int s = 0; s < 24; ++s) {
    int task = wave + 8 * s;  // 0..191
    if (task < 128) {
      int proj = task >> 6;  // 0=q, 1=k
      int t2 = task & 63;
      int nt = t2 & 3, dt = t2 >> 2;
      const _Float16* w16 = proj ? w16k : w16q;
      const float* bias = proj ? bk : bq;
      _Float16* outp = proj ? kT : qT;
      float bval = bias[dt * 16 + m16];
      v8f acc;
#pragma unroll
      for (int r = 0; r < 8; ++r) acc[r] = bval;
#pragma unroll
      for (int cc = 0; cc < 8; ++cc) {
        v16h a = load_frag(&xT[nt * 16 + m16][cc * 32], hk);
        v16h bb = load_frag(w16 + (size_t)(dt * 16 + m16) * C + cc * 32, hk);
        acc = wmma_f16(a, bb, acc);
      }
#pragma unroll
      for (int r = 0; r < 8; ++r) {
        int i_abs = n0 + nt * 16 + r + 8 * hk;
        outp[((size_t)b * N + i_abs) * C + dt * 16 + m16] = (_Float16)acc[r];
      }
    } else {
      int t2 = task - 128;
      int nt = t2 & 3, dt = t2 >> 2;
      v8f acc;
#pragma unroll
      for (int r = 0; r < 8; ++r) acc[r] = bv[dt * 16 + r + 8 * hk];
#pragma unroll
      for (int cc = 0; cc < 8; ++cc) {
        v16h a = load_frag(w16v + (size_t)(dt * 16 + m16) * C + cc * 32, hk);
        v16h bb = load_frag(&xT[nt * 16 + m16][cc * 32], hk);
        acc = wmma_f16(a, bb, acc);
      }
#pragma unroll
      for (int r = 0; r < 8; ++r) {
        int d_abs = dt * 16 + r + 8 * hk;
        v16[((size_t)b * C + d_abs) * N + n0 + nt * 16 + m16] = (_Float16)acc[r];
      }
    }
  }
}

// ---------------- Kernel 2: fused flash attention + residual ----------------
// 8 waves x 16 query rows. K/V tiles double-buffered in LDS; on CDNA5 the tiles are
// fetched by the Tensor Data Mover (wave 0 issues descriptors, TENSORcnt + barrier
// hand the buffer to the workgroup), overlapping DMA with WMMA compute.
__global__ __launch_bounds__(THREADS) void k_attn(
    const float* __restrict__ pose, const float* __restrict__ gamma,
    const _Float16* __restrict__ qT, const _Float16* __restrict__ kT,
    const _Float16* __restrict__ v16, float* __restrict__ out) {
  extern __shared__ __align__(16) _Float16 smem[];
  _Float16* kbuf = smem + KBUF_E;  // [2][JB][C]
  _Float16* vbuf = smem + VBUF_E;  // [2][C][JB]
  const int tid = threadIdx.x;
  const int b = blockIdx.x / (N / 128);
  const int i0 = (blockIdx.x % (N / 128)) * 128 + (tid >> 5) * 16;
  const int wave = tid >> 5, lane = tid & 31, m16 = lane & 15, hk = lane >> 4;
  _Float16* pb = smem + PBUF_E + wave * 16 * JB;  // per-wave P relayout scratch
  const float g = gamma[0];

  const _Float16* kbase = kT + (size_t)b * N * C;
  const _Float16* vbase = v16 + (size_t)b * C * N;

  // Q fragments for this wave's 16 rows, kept in registers
  v16h qa[8];
  const _Float16* qrow = qT + ((size_t)b * N + i0 + m16) * C;
#pragma unroll
  for (int cc = 0; cc < 8; ++cc) qa[cc] = load_frag(qrow + cc * 32, hk);

  v8f O[16];
  v8f z = {0.f, 0.f, 0.f, 0.f, 0.f, 0.f, 0.f, 0.f};
#pragma unroll
  for (int ct = 0; ct < 16; ++ct) O[ct] = z;
  float m8[8], l8[8];
#pragma unroll
  for (int r = 0; r < 8; ++r) { m8[r] = -3.0e38f; l8[r] = 0.f; }

  const int NB = N / JB;

  // ---- prologue: fetch tile 0 into buffer 0 ----
#if defined(HAVE_TDM)
  if (wave == 0) {
    tdm_load_2d(KBUF_E * 2, kbase, C, N, C, JB, C);
    tdm_load_2d(VBUF_E * 2, vbase, N, C, JB, C, N);
    __builtin_amdgcn_s_wait_tensorcnt(0);
  }
#else
  {
#pragma unroll
    for (int t = 0; t < 4; ++t) {
      int id = tid + THREADS * t;
      int row = id >> 5, ch = id & 31;
      *(v8h*)(kbuf + row * C + ch * 8) = *(const v8h*)(kbase + (size_t)row * C + ch * 8);
    }
#pragma unroll
    for (int t = 0; t < 4; ++t) {
      int id = tid + THREADS * t;
      int c = id >> 2, ch = id & 3;
      *(v8h*)(vbuf + c * JB + ch * 8) = *(const v8h*)(vbase + (size_t)c * N + ch * 8);
    }
  }
#endif
  __syncthreads();

  for (int jb = 0; jb < NB; ++jb) {
    const int cur = jb & 1;
    const int j0 = jb * JB;
    const _Float16* kb = kbuf + cur * JB * C;
    const _Float16* vb = vbuf + cur * C * JB;

    // ---- fetch next tile into the other buffer while computing this one ----
    if (jb + 1 < NB) {
#if defined(HAVE_TDM)
      if (wave == 0) {
        tdm_load_2d((KBUF_E + (1 - cur) * JB * C) * 2, kbase + (size_t)(j0 + JB) * C, C, N, C, JB,
                    C);
        tdm_load_2d((VBUF_E + (1 - cur) * C * JB) * 2, vbase + (j0 + JB), N, C, JB, C, N);
      }
#else
      _Float16* kn = kbuf + (1 - cur) * JB * C;
      _Float16* vn = vbuf + (1 - cur) * C * JB;
#pragma unroll
      for (int t = 0; t < 4; ++t) {
        int id = tid + THREADS * t;
        int row = id >> 5, ch = id & 31;
        *(v8h*)(kn + row * C + ch * 8) =
            *(const v8h*)(kbase + (size_t)(j0 + JB + row) * C + ch * 8);
      }
#pragma unroll
      for (int t = 0; t < 4; ++t) {
        int id = tid + THREADS * t;
        int c = id >> 2, ch = id & 3;
        *(v8h*)(vn + c * JB + ch * 8) = *(const v8h*)(vbase + (size_t)c * N + j0 + JB + ch * 8);
      }
#endif
    }
    if (jb + 2 < NB) {  // prefetch two blocks ahead into L2 (global_prefetch_b8)
      __builtin_prefetch(kbase + (size_t)(j0 + 2 * JB + (tid >> 3)) * C + (tid & 7) * 32, 0, 0);
      __builtin_prefetch(vbase + (size_t)tid * N + j0 + 2 * JB, 0, 0);
    }

    // ---- S = Q^T K for this j block ----
    v8f S[2];
    S[0] = z;
    S[1] = z;
#pragma unroll
    for (int jt = 0; jt < 2; ++jt) {
#pragma unroll
      for (int cc = 0; cc < 8; ++cc) {
        v16h kf = load_frag(kb + (size_t)(jt * 16 + m16) * C + cc * 32, hk);
        S[jt] = wmma_f16(qa[cc], kf, S[jt]);
      }
    }

    // ---- online softmax (rows r+8hk per lane-half) ----
    float rm[8];
#pragma unroll
    for (int r = 0; r < 8; ++r) rm[r] = fmaxf(S[0][r], S[1][r]);
#pragma unroll
    for (int mask = 1; mask < 16; mask <<= 1) {
#pragma unroll
      for (int r = 0; r < 8; ++r) rm[r] = fmaxf(rm[r], __shfl_xor(rm[r], mask, 32));
    }
    float alpha[8];
#pragma unroll
    for (int r = 0; r < 8; ++r) {
      float mn = fmaxf(m8[r], rm[r]);
      alpha[r] = __expf(m8[r] - mn);
      m8[r] = mn;
    }
#pragma unroll
    for (int jt = 0; jt < 2; ++jt) {
#pragma unroll
      for (int r = 0; r < 8; ++r) S[jt][r] = __expf(S[jt][r] - m8[r]);
    }
    float rs[8];
#pragma unroll
    for (int r = 0; r < 8; ++r) rs[r] = S[0][r] + S[1][r];
#pragma unroll
    for (int mask = 1; mask < 16; mask <<= 1) {
#pragma unroll
      for (int r = 0; r < 8; ++r) rs[r] += __shfl_xor(rs[r], mask, 32);
    }
#pragma unroll
    for (int r = 0; r < 8; ++r) l8[r] = l8[r] * alpha[r] + rs[r];
#pragma unroll
    for (int ct = 0; ct < 16; ++ct) {
#pragma unroll
      for (int r = 0; r < 8; ++r) O[ct][r] *= alpha[r];
    }

    // ---- relayout P (C-layout f32) -> A fragment f16 via per-wave LDS scratch ----
#pragma unroll
    for (int jt = 0; jt < 2; ++jt) {
#pragma unroll
      for (int r = 0; r < 8; ++r) pb[(r + 8 * hk) * JB + jt * 16 + m16] = (_Float16)S[jt][r];
    }
    asm volatile("s_wait_dscnt 0" ::: "memory");  // same-wave LDS RAW fence
    v16h pa = load_frag(pb + m16 * JB, hk);

    // ---- O += P * V^T (A operand identical across the chain -> reuse_a hint on 1..15) ----
    {
      v16h vf0 = load_frag(vb + (size_t)m16 * JB, hk);
      O[0] = wmma_f16_t<false>(pa, vf0, O[0]);
    }
#pragma unroll
    for (int ct = 1; ct < 16; ++ct) {
      v16h vf = load_frag(vb + (size_t)(ct * 16 + m16) * JB, hk);
      O[ct] = wmma_f16_t<true>(pa, vf, O[ct]);
    }

    // ---- hand next buffer to the workgroup ----
    if (jb + 1 < NB) {
#if defined(HAVE_TDM)
      if (wave == 0) __builtin_amdgcn_s_wait_tensorcnt(0);
#endif
      __syncthreads();
    }
  }

  // ---- epilogue: out[b,c,i] = pose + gamma * O / l (rows address-contiguous -> float4) ----
#pragma unroll
  for (int r = 0; r < 8; ++r) l8[r] = g / l8[r];
#pragma unroll
  for (int ct = 0; ct < 16; ++ct) {
    int c_abs = ct * 16 + m16;
    size_t base = ((size_t)b * C + c_abs) * N + i0 + 8 * hk;
    const float* pp = pose + base;
    float* op = out + base;
    float4 p0 = *(const float4*)(pp);
    float4 p1 = *(const float4*)(pp + 4);
    float4 o0, o1;
    o0.x = p0.x + O[ct][0] * l8[0];
    o0.y = p0.y + O[ct][1] * l8[1];
    o0.z = p0.z + O[ct][2] * l8[2];
    o0.w = p0.w + O[ct][3] * l8[3];
    o1.x = p1.x + O[ct][4] * l8[4];
    o1.y = p1.y + O[ct][5] * l8[5];
    o1.z = p1.z + O[ct][6] * l8[6];
    o1.w = p1.w + O[ct][7] * l8[7];
    *(float4*)(op) = o0;
    *(float4*)(op + 4) = o1;
  }
}

extern "C" void kernel_launch(void* const* d_in, const int* in_sizes, int n_in,
                              void* d_out, int out_size, void* d_ws, size_t ws_size,
                              hipStream_t stream) {
  const float* pose = (const float*)d_in[0];
  const float* wq = (const float*)d_in[1];
  const float* bq = (const float*)d_in[2];
  const float* wk = (const float*)d_in[3];
  const float* bk = (const float*)d_in[4];
  const float* wv = (const float*)d_in[5];
  const float* bv = (const float*)d_in[6];
  const float* gamma = (const float*)d_in[7];
  char* ws = (char*)d_ws;
  _Float16* w16 = (_Float16*)(ws + W16Q_OFF);
  _Float16* w16q = (_Float16*)(ws + W16Q_OFF);
  _Float16* w16k = (_Float16*)(ws + W16K_OFF);
  _Float16* w16v = (_Float16*)(ws + W16V_OFF);
  _Float16* qT = (_Float16*)(ws + QT16_OFF);
  _Float16* kT = (_Float16*)(ws + KT16_OFF);
  _Float16* v16p = (_Float16*)(ws + V16_OFF);
  float* out = (float*)d_out;

  (void)hipFuncSetAttribute((const void*)k_attn, hipFuncAttributeMaxDynamicSharedMemorySize,
                            (int)SMEM_BYTES);

  k_cvt_w<<<(3 * C * C + THREADS - 1) / THREADS, THREADS, 0, stream>>>(wq, wk, wv, w16);
  k_qkv<<<dim3(B * (N / 64)), THREADS, 0, stream>>>(pose, bq, bk, bv, w16q, w16k, w16v, qT, kT,
                                                    v16p);
  k_attn<<<dim3(B * (N / 128)), THREADS, SMEM_BYTES, stream>>>(pose, gamma, qT, kT, v16p, out);
}
